// MoEBlock_7241314861577
// MI455X (gfx1250) — compile-verified, matrix operand order checked by reading, not run on hardware
//
#include <hip/hip_runtime.h>
#include <hip/hip_bf16.h>
#include <stdint.h>

// ---------------- problem constants ----------------
#define DMODEL 1024
#define HDIM   4096
#define NEXP   8
#define NTOK   8192      // B*S = 4*2048
#define NPAIR  16384     // NTOK * K
#define CAPN   2560      // capacity per expert

// ---------------- types ----------------
typedef __attribute__((ext_vector_type(16))) __bf16 bf16x16;
typedef __attribute__((ext_vector_type(8)))  __bf16 bf16x8;
typedef __attribute__((ext_vector_type(4)))  __bf16 bf16x4;
typedef __attribute__((ext_vector_type(8)))  float  v8f;
typedef __attribute__((ext_vector_type(4)))  unsigned int v4u;
typedef __attribute__((ext_vector_type(4)))  int v4i;
typedef __attribute__((ext_vector_type(8)))  int v8i;

union BF16Frag { bf16x16 v; bf16x8 h[2]; };

// ---- CDNA5 TDM availability (probe-based; 5-arg ROCm7.2 vs 6-arg therock) ----
#if __has_builtin(__builtin_amdgcn_tensor_load_to_lds) && \
    __has_builtin(__builtin_amdgcn_s_wait_tensorcnt)
  #define USE_TDM 1
  #if __has_include(<hip/amd_detail/amd_gfx1250_TDM.h>)
    #define TDM_6ARG 1
  #endif
#else
  #define USE_TDM 0
#endif

__device__ inline unsigned ballot32(bool p) {
#if __has_builtin(__builtin_amdgcn_ballot_w32)
  return __builtin_amdgcn_ballot_w32(p);
#else
  return (unsigned)__ballot(p);
#endif
}

// tanh-approx GELU in sigmoid form: one v_exp_f32 instead of a libm tanh chain.
// 0.5*x*(1+tanh(v)) == x * sigmoid(2v),  2v = 2*sqrt(2/pi)*(x + 0.044715 x^3)
__device__ inline float gelu_tanh(float x) {
  float u = 1.5957691216057308f * (x + 0.044715f * x * x * x);
  return x / (1.0f + __expf(-u));
}

#if USE_TDM
// Issue one 2D TDM tile load: 32(dim0) x 128(dim1) bf16 elements, row stride
// `strideElems`, into LDS at `ldsOff` with 8-bf16 padding per 32-elem row
// (pad_interval: 16 DWORDs -> code 3, pad_amount: 4 DWORDs -> code 3),
// reproducing the [128][40] padded LDS tile layout.
__device__ inline void tdm_load_2d(unsigned ldsOff, const void* gptr,
                                   int tensorD0, int tensorD1, int strideElems) {
  unsigned long long ga = (unsigned long long)(uintptr_t)gptr;
  v4u g0;
  g0[0] = 1u;                                              // count=1 (valid D#)
  g0[1] = ldsOff;                                          // lds_addr
  g0[2] = (unsigned)(ga & 0xffffffffu);                    // global_addr[31:0]
  g0[3] = (unsigned)((ga >> 32) & 0x01ffffffu) | (2u << 30);  // addr[56:32] | type=2
  v8i g1;
  g1[0] = (1 << 16) | (1 << 20) | (3 << 22) | (3 << 25);   // data_size=2B, pad en, 16dw interval, 4dw pad
  g1[1] = (int)((unsigned)(tensorD0 & 0xffff) << 16);      // tensor_dim0[15:0]
  g1[2] = (int)(((unsigned)tensorD0 >> 16) | ((unsigned)(tensorD1 & 0xffff) << 16));
  g1[3] = (int)(((unsigned)tensorD1 >> 16) | (32u << 16)); // tile_dim0 = 32
  g1[4] = 128;                                             // tile_dim1 = 128
  g1[5] = strideElems;                                     // dim0_stride[31:0]
  g1[6] = 0;
  g1[7] = 0;
  v4i z4 = {0, 0, 0, 0};
#if defined(TDM_6ARG)
  v8i z8 = {0, 0, 0, 0, 0, 0, 0, 0};
  __builtin_amdgcn_tensor_load_to_lds(g0, g1, z4, z4, z8, 0);
#else
  __builtin_amdgcn_tensor_load_to_lds(g0, g1, z4, z4, 0);
#endif
}
#endif

// ============ 0) Weight transpose + fp32->bf16 convert ============
// src: [E][R][C] fp32 row-major ; dst: [E][C][R] bf16 row-major
__global__ __launch_bounds__(256) void transpose_cvt_kernel(
    const float* __restrict__ src, unsigned short* __restrict__ dst_,
    int R, int C) {
  __bf16* dst = (__bf16*)dst_;
  __shared__ float tile[32][33];
  const int e  = blockIdx.z;
  const int r0 = blockIdx.y * 32;
  const int c0 = blockIdx.x * 32;
  const int tx = threadIdx.x & 31, ty = threadIdx.x >> 5;  // 32 x 8
  const float* s = src + (size_t)e * R * C;
  __bf16* d = dst + (size_t)e * C * R;
#pragma unroll
  for (int i = 0; i < 4; i++)
    tile[ty + 8 * i][tx] = s[(size_t)(r0 + ty + 8 * i) * C + c0 + tx];
  __syncthreads();
#pragma unroll
  for (int i = 0; i < 4; i++)
    d[(size_t)(c0 + ty + 8 * i) * R + r0 + tx] = (__bf16)tile[tx][ty + 8 * i];
}

// ============ 1) Router: logits -> top2 + renormalized weights ============
__global__ __launch_bounds__(256) void router_kernel(
    const float* __restrict__ x, const float* __restrict__ Wr,
    int* __restrict__ eids, float* __restrict__ wts) {
  __shared__ float WrS[DMODEL * NEXP];           // 32 KB
  const int tid = threadIdx.x;
  for (int i = tid; i < DMODEL * NEXP; i += 256) WrS[i] = Wr[i];
  __syncthreads();

  const int lane = tid & 31, wid = tid >> 5;
  const int t = blockIdx.x * 8 + wid;            // one wave per token
  const float* xr = x + (size_t)t * DMODEL;

  float acc[NEXP];
#pragma unroll
  for (int e = 0; e < NEXP; e++) acc[e] = 0.0f;

  for (int d = lane; d < DMODEL; d += 32) {
    float xv = xr[d];
#pragma unroll
    for (int e = 0; e < NEXP; e++) acc[e] += xv * WrS[d * NEXP + e];
  }
#pragma unroll
  for (int off = 16; off >= 1; off >>= 1) {
#pragma unroll
    for (int e = 0; e < NEXP; e++) acc[e] += __shfl_down(acc[e], off, 32);
  }
  if (lane == 0) {
    int i1 = -1, i2 = -1;
    float l1 = -3.4e38f, l2 = -3.4e38f;
#pragma unroll
    for (int e = 0; e < NEXP; e++) {      // strict '>' => lower index wins ties
      float v = acc[e];
      if (v > l1) { l2 = l1; i2 = i1; l1 = v; i1 = e; }
      else if (v > l2) { l2 = v; i2 = e; }
    }
    float r  = __expf(l2 - l1);
    float w1 = 1.0f / (1.0f + r);
    eids[2 * t]     = i1;  eids[2 * t + 1] = i2;
    wts[2 * t]      = w1;  wts[2 * t + 1]  = 1.0f - w1;
  }
}

// ===== 2) Ordered per-expert cumulative positions (matches jnp.cumsum order) =====
__global__ __launch_bounds__(256) void pos_kernel(
    const int* __restrict__ eids, int* __restrict__ dest) {
  __shared__ int runBase[NEXP];
  __shared__ int waveHist[8][NEXP];
  const int tid = threadIdx.x, lane = tid & 31, wid = tid >> 5;
  if (tid < NEXP) runBase[tid] = 0;
  __syncthreads();

  for (int chunk = 0; chunk < NPAIR / 256; chunk++) {
    const int p = chunk * 256 + tid;
    const int e = eids[p];
    unsigned mine = 0;
#pragma unroll
    for (int q = 0; q < NEXP; q++) {
      unsigned m = ballot32(e == q);
      if (e == q) mine = m;
      if (lane == 0) waveHist[wid][q] = __popc(m);
    }
    const int waveRank = __popc(mine & ((1u << lane) - 1u));
    __syncthreads();
    int base = runBase[e];
    for (int w = 0; w < wid; w++) base += waveHist[w][e];
    const int pos = base + waveRank;
    dest[p] = (pos < CAPN) ? (e * CAPN + pos) : -1;
    __syncthreads();
    if (tid < NEXP) {
      int s = 0;
#pragma unroll
      for (int w = 0; w < 8; w++) s += waveHist[w][tid];
      runBase[tid] += s;
    }
    __syncthreads();
  }
}

// ============ 3) Scatter kept pairs into expert_in (bf16) ============
__global__ __launch_bounds__(256) void scatter_kernel(
    const float* __restrict__ x, const int* __restrict__ dest,
    unsigned short* __restrict__ Xe_) {
  __bf16* Xe = (__bf16*)Xe_;
  const int p = blockIdx.x;
  const int d = dest[p];
  if (d < 0) return;
  const int t = p >> 1;
  const int idx = threadIdx.x * 4;
  float4 v = *(const float4*)(x + (size_t)t * DMODEL + idx);
  bf16x4 o;
  o[0] = (__bf16)v.x; o[1] = (__bf16)v.y; o[2] = (__bf16)v.z; o[3] = (__bf16)v.w;
  *(bf16x4*)(Xe + (size_t)d * DMODEL + idx) = o;
}

// ============ 4/5) Grouped expert GEMM: C = A(bf16) x Bt(bf16, N-major) + bias ============
// A:  [E, CAPN, KDIM] bf16 ; Bt: [E, NDIM, KDIM] bf16 (pre-transposed weights)
// block tile 128(M) x 128(N), K step 32, 8 waves each computing 32x64.
template <int KDIM, int NDIM, bool GELU>
__global__ __launch_bounds__(256) void moe_gemm_bf16(
    const unsigned short* __restrict__ A_,
    const unsigned short* __restrict__ Bt_,
    const float* __restrict__ bias,          // [E, NDIM]
    unsigned short* __restrict__ outBf_,     // used if GELU  (bf16)
    float* __restrict__ outF) {              // used if !GELU
  const __bf16* A  = (const __bf16*)A_;
  const __bf16* Bt = (const __bf16*)Bt_;
  __bf16* outBf = (__bf16*)outBf_;

  const int e  = blockIdx.z;
  const int mt = blockIdx.y;
  const int nt = blockIdx.x;
  const int tid  = threadIdx.x;
  const int lane = tid & 31, wid = tid >> 5;
  const int waveM = wid >> 1, waveN = wid & 1;   // 4x2 wave grid
  const int lr = lane & 15;
  const int hb = lane >> 4;
  const int kb = hb * 8;

  const __bf16* Ae = A  + (size_t)e * CAPN * KDIM + (size_t)(mt * 128) * KDIM;
  const __bf16* Be = Bt + (size_t)e * NDIM * KDIM + (size_t)(nt * 128) * KDIM;

  __shared__ __align__(16) __bf16 As[2][128][40];   // [m][k], rows padded to 40
  __shared__ __align__(16) __bf16 Bs[2][128][40];   // [n][k], rows padded to 40

  v8f acc[2][4];
  const v8f vzero = {0.f, 0.f, 0.f, 0.f, 0.f, 0.f, 0.f, 0.f};
#pragma unroll
  for (int i = 0; i < 2; i++)
#pragma unroll
    for (int j = 0; j < 4; j++) acc[i][j] = vzero;

  auto compute = [&](int buf) {
    bf16x16 a[2], b[4];
#pragma unroll
    for (int i = 0; i < 2; i++) {
      int r = waveM * 32 + i * 16 + lr;
      BF16Frag u;
      u.h[0] = *(const bf16x8*)&As[buf][r][kb];        // K = kb..kb+7
      u.h[1] = *(const bf16x8*)&As[buf][r][kb + 16];   // K = kb+16..kb+23
      a[i] = u.v;
    }
#pragma unroll
    for (int j = 0; j < 4; j++) {
      int n = waveN * 64 + j * 16 + lr;
      BF16Frag u;
      u.h[0] = *(const bf16x8*)&Bs[buf][n][kb];
      u.h[1] = *(const bf16x8*)&Bs[buf][n][kb + 16];
      b[j] = u.v;
    }
#pragma unroll
    for (int i = 0; i < 2; i++)
#pragma unroll
      for (int j = 0; j < 4; j++)
        acc[i][j] = __builtin_amdgcn_wmma_f32_16x16x32_bf16(
            false, a[i], false, b[j], (short)0, acc[i][j], false, false);
  };

  constexpr int NK = KDIM / 32;
  static_assert((NK & 1) == 0, "pipeline unrolled by 2");

#if USE_TDM
  // ---- Tensor Data Mover pipeline: wave 0 drives the DMA, all waves compute ----
  // Unrolled by 2 so both LDS buffer addresses are compile-time constants.
  const bool issuer = (wid == 0);
  const unsigned ldsA0 = (unsigned)(uintptr_t)&As[0][0][0];
  const unsigned ldsA1 = (unsigned)(uintptr_t)&As[1][0][0];
  const unsigned ldsB0 = (unsigned)(uintptr_t)&Bs[0][0][0];
  const unsigned ldsB1 = (unsigned)(uintptr_t)&Bs[1][0][0];
  if (issuer) {
    tdm_load_2d(ldsA0, Ae, KDIM, CAPN, KDIM);
    tdm_load_2d(ldsB0, Be, KDIM, NDIM, KDIM);
  }
#pragma unroll 1
  for (int kt = 0; kt < NK; kt += 2) {
    // ---- even stage: compute buffer 0, TDM fills buffer 1 ----
    if (issuer) {
      tdm_load_2d(ldsA1, Ae + (size_t)(kt + 1) * 32, KDIM, CAPN, KDIM);
      tdm_load_2d(ldsB1, Be + (size_t)(kt + 1) * 32, KDIM, NDIM, KDIM);
      __builtin_amdgcn_s_wait_tensorcnt(2);   // stage kt landed, kt+1 in flight
    }
    __syncthreads();
    compute(0);
    __syncthreads();
    // ---- odd stage: compute buffer 1, TDM fills buffer 0 ----
    if (issuer) {
      if (kt + 2 < NK) {
        tdm_load_2d(ldsA0, Ae + (size_t)(kt + 2) * 32, KDIM, CAPN, KDIM);
        tdm_load_2d(ldsB0, Be + (size_t)(kt + 2) * 32, KDIM, NDIM, KDIM);
        __builtin_amdgcn_s_wait_tensorcnt(2);
      } else {
        __builtin_amdgcn_s_wait_tensorcnt(0);
      }
    }
    __syncthreads();
    compute(1);
    __syncthreads();
  }
#else
  // ---- fallback: manual double-buffered global->reg->LDS pipeline ----
  bf16x4 aReg[4], bReg[4];
  auto loadG = [&](int kt) {
    const int k0 = kt * 32;
#pragma unroll
    for (int j = 0; j < 4; j++) {
      int c = tid + 256 * j;                 // 0..1023
      int r  = c >> 3;                       // 128 rows x 8 chunks of 4 bf16
      int c4 = (c & 7) * 4;
      aReg[j] = *(const bf16x4*)(Ae + (size_t)r * KDIM + k0 + c4);
      bReg[j] = *(const bf16x4*)(Be + (size_t)r * KDIM + k0 + c4);
    }
  };
  auto storeS = [&](int buf) {
#pragma unroll
    for (int j = 0; j < 4; j++) {
      int c = tid + 256 * j;
      int r  = c >> 3;
      int c4 = (c & 7) * 4;
      *(bf16x4*)&As[buf][r][c4] = aReg[j];
      *(bf16x4*)&Bs[buf][r][c4] = bReg[j];
    }
  };
  loadG(0);
  storeS(0);
  __syncthreads();
  int buf = 0;
#pragma unroll 1
  for (int kt = 0; kt < NK; kt++) {
    if (kt + 2 < NK)
      __builtin_prefetch(Be + (size_t)(tid >> 3) * KDIM + (kt + 2) * 32, 0, 0);
    if (kt + 1 < NK) loadG(kt + 1);
    compute(buf);
    if (kt + 1 < NK) storeS(buf ^ 1);
    __syncthreads();
    buf ^= 1;
  }
#endif

  // epilogue — ISA C/D layout: lanes 0-15 hold M=v, lanes 16-31 hold M=8+v; N=lane&15
#pragma unroll
  for (int i = 0; i < 2; i++) {
    const int mbase = mt * 128 + waveM * 32 + i * 16 + hb * 8;
#pragma unroll
    for (int j = 0; j < 4; j++) {
      const int n = nt * 128 + waveN * 64 + j * 16 + lr;
      const float bv = bias[(size_t)e * NDIM + n];
#pragma unroll
      for (int v = 0; v < 8; v++) {
        const int m = mbase + v;
        float val = acc[i][j][v] + bv;
        if (GELU) {
          outBf[((size_t)e * CAPN + m) * NDIM + n] = (__bf16)gelu_tanh(val);
        } else {
          outF[((size_t)e * CAPN + m) * NDIM + n] = val;
        }
      }
    }
  }
}

// ============ 6) Combine: y[t] = w0*out[d0] + w1*out[d1] ============
__global__ __launch_bounds__(256) void combine_kernel(
    const float* __restrict__ Eout, const int* __restrict__ dest,
    const float* __restrict__ wts, float* __restrict__ out) {
  const int t = blockIdx.x;
  const int idx = threadIdx.x * 4;
  const int d0 = dest[2 * t], d1 = dest[2 * t + 1];
  const float w0 = wts[2 * t], w1 = wts[2 * t + 1];
  float4 r = {0.f, 0.f, 0.f, 0.f};
  if (d0 >= 0) {
    float4 v = *(const float4*)(Eout + (size_t)d0 * DMODEL + idx);
    r.x += w0 * v.x; r.y += w0 * v.y; r.z += w0 * v.z; r.w += w0 * v.w;
  }
  if (d1 >= 0) {
    float4 v = *(const float4*)(Eout + (size_t)d1 * DMODEL + idx);
    r.x += w1 * v.x; r.y += w1 * v.y; r.z += w1 * v.z; r.w += w1 * v.w;
  }
  *(float4*)(out + (size_t)t * DMODEL + idx) = r;
}

// ---------------- launcher ----------------
extern "C" void kernel_launch(void* const* d_in, const int* in_sizes, int n_in,
                              void* d_out, int out_size, void* d_ws, size_t ws_size,
                              hipStream_t stream) {
  const float* x  = (const float*)d_in[0];
  const float* Wr = (const float*)d_in[1];
  const float* W1 = (const float*)d_in[2];
  const float* b1 = (const float*)d_in[3];
  const float* W2 = (const float*)d_in[4];
  const float* b2 = (const float*)d_in[5];
  float* out = (float*)d_out;

  char* ws = (char*)d_ws;
  int*   eids = (int*)(ws);                                   // 64 KB
  int*   dest = (int*)(ws + 65536);                           // 64 KB
  float* wts  = (float*)(ws + 131072);                        // 64 KB
  unsigned short* Xe  = (unsigned short*)(ws + 196608);                          // 40 MB  bf16 [E,CAP,D]
  unsigned short* Hb  = (unsigned short*)(ws + 196608 + 41943040ull);            // 160 MB bf16 [E,CAP,H]
  float*          Eout = (float*)(ws + 196608 + 41943040ull + 167772160ull);     // 80 MB  f32  [E,CAP,D]
  unsigned short* Wt1 = (unsigned short*)(ws + 196608 + 41943040ull + 167772160ull + 83886080ull);           // 64 MB bf16 [E,H,D]
  unsigned short* Wt2 = (unsigned short*)(ws + 196608 + 41943040ull + 167772160ull + 83886080ull + 67108864ull); // 64 MB bf16 [E,D,H]

  // weights -> bf16, N-major (B matrices become same layout as A)
  transpose_cvt_kernel<<<dim3(HDIM / 32, DMODEL / 32, NEXP), 256, 0, stream>>>(
      W1, Wt1, DMODEL, HDIM);
  transpose_cvt_kernel<<<dim3(DMODEL / 32, HDIM / 32, NEXP), 256, 0, stream>>>(
      W2, Wt2, HDIM, DMODEL);

  router_kernel<<<NTOK / 8, 256, 0, stream>>>(x, Wr, eids, wts);
  pos_kernel<<<1, 256, 0, stream>>>(eids, dest);
  scatter_kernel<<<NPAIR, 256, 0, stream>>>(x, dest, Xe);

  moe_gemm_bf16<DMODEL, HDIM, true>
      <<<dim3(HDIM / 128, CAPN / 128, NEXP), 256, 0, stream>>>(Xe, Wt1, b1, Hb, nullptr);
  moe_gemm_bf16<HDIM, DMODEL, false>
      <<<dim3(DMODEL / 128, CAPN / 128, NEXP), 256, 0, stream>>>(Hb, Wt2, b2, nullptr, Eout);

  combine_kernel<<<NTOK, 256, 0, stream>>>(Eout, dest, wts, out);
}
